// RenderXY_31593779429763
// MI455X (gfx1250) — matrix operand order repaired and checked by reading, not compiled, not run on hardware
//
#include <hip/hip_runtime.h>

namespace {

constexpr int   kXd      = 1024;
constexpr int   kQ       = 32;
constexpr int   kHid     = 32;
constexpr int   kNPix    = 2048 * 2048;
constexpr int   kBlock   = 256;
constexpr int   kWaves   = kBlock / 32;
constexpr int   kAStride = 40;            // halfs per A-tile row (32 data + 8 pad)
constexpr int   kWStride = 40;            // halfs per transposed-weight row
constexpr float kScale    = 1024.0f;      // feature pre-scale: keep f16 out of denormals
constexpr float kInvScale = 1.0f / 1024.0f;

typedef _Float16 half_t;
typedef __attribute__((ext_vector_type(4)))  _Float16 v4h;
typedef __attribute__((ext_vector_type(8)))  _Float16 v8h;
typedef __attribute__((ext_vector_type(16))) _Float16 v16h;
typedef __attribute__((ext_vector_type(4)))  float    v4f;
typedef __attribute__((ext_vector_type(8)))  float    v8f;

__global__ __launch_bounds__(kBlock)
void render_mlp_kernel(const float* __restrict__ M,
                       const int* __restrict__ x0, const int* __restrict__ y0,
                       const int* __restrict__ x1, const int* __restrict__ y1,
                       const float* __restrict__ wx, const float* __restrict__ wy,
                       const float* __restrict__ W1, const float* __restrict__ b1,
                       const float* __restrict__ W2, const float* __restrict__ b2,
                       const float* __restrict__ W3, const float* __restrict__ b3,
                       float* __restrict__ out)
{
    __shared__ half_t sW1T[kHid * kWStride];
    __shared__ half_t sW2T[kHid * kWStride];
    __shared__ half_t sA[kWaves * 32 * kAStride];

    const int tid  = threadIdx.x;
    const int wave = tid >> 5;
    const int lane = tid & 31;
    const int l16  = lane & 15;   // lane within 16-lane half
    const int hsel = lane >> 4;   // which half of the wave

    // ---- stage weights transposed as f16 (WT[n][k]) into padded LDS rows ----
    for (int i = tid; i < kQ * kHid; i += kBlock) {
        const int k = i >> 5;
        const int n = i & 31;
        sW1T[n * kWStride + k] = (half_t)W1[k * kHid + n];
        sW2T[n * kWStride + k] = (half_t)W2[k * kHid + n];
    }
    __syncthreads();

    const int pixBase = blockIdx.x * kBlock + wave * 32;
    const int pix     = pixBase + lane;

    // ---- bilinear gather -> pre-scaled f16 A-tile row in LDS ----
    {
        const float fwx = wx[pix];
        const float fwy = wy[pix];
        const float w00 = (1.0f - fwx) * (1.0f - fwy) * kScale;
        const float w10 = fwx * (1.0f - fwy) * kScale;
        const float w01 = (1.0f - fwx) * fwy * kScale;
        const float w11 = fwx * fwy * kScale;
        const float* p00 = M + (y0[pix] * kXd + x0[pix]) * kQ;
        const float* p10 = M + (y0[pix] * kXd + x1[pix]) * kQ;
        const float* p01 = M + (y1[pix] * kXd + x0[pix]) * kQ;
        const float* p11 = M + (y1[pix] * kXd + x1[pix]) * kQ;
        half_t* row = &sA[(wave * 32 + lane) * kAStride];
#pragma unroll
        for (int q = 0; q < kQ; q += 4) {
            const v4f a = *(const v4f*)(p00 + q);
            const v4f b = *(const v4f*)(p10 + q);
            const v4f c = *(const v4f*)(p01 + q);
            const v4f d = *(const v4f*)(p11 + q);
            const v4f f = a * w00 + b * w10 + c * w01 + d * w11;
            v4h h;
            h.x = (half_t)f.x; h.y = (half_t)f.y;
            h.z = (half_t)f.z; h.w = (half_t)f.w;
            *(v4h*)(row + q) = h;
        }
    }
    __syncthreads();

    const half_t* aBase = &sA[wave * 32 * kAStride];

    // 16-bit A layout: lane m = L%16; VGPR0-3 hold K = (0..7)+8*hsel, VGPR4-7 hold K = 16..23 + 8*hsel
    auto loadA = [&](int mt) -> v16h {
        const half_t* p = aBase + (mt * 16 + l16) * kAStride + hsel * 8;
        const v8h lo = *(const v8h*)(p);
        const v8h hi = *(const v8h*)(p + 16);
        v16h r;
#pragma unroll
        for (int i = 0; i < 8; ++i) { r[i] = lo[i]; r[i + 8] = hi[i]; }
        return r;
    };
    // 16-bit B layout: N = L%16; VGPR0-3 hold K = (0..7)+8*hsel, VGPR4-7 hold K = 16..23 + 8*hsel
    auto loadB = [&](const half_t* WT, int nt) -> v16h {
        const half_t* p = WT + (nt * 16 + l16) * kWStride + hsel * 8;
        const v8h lo = *(const v8h*)(p);
        const v8h hi = *(const v8h*)(p + 16);
        v16h r;
#pragma unroll
        for (int i = 0; i < 8; ++i) { r[i] = lo[i]; r[i + 8] = hi[i]; }
        return r;
    };

    v8f acc[2][2];

    // ---- layer 1: [32 pixels x 32] @ [32 x 32] via 4 WMMAs ----
    {
        const v16h B0 = loadB(sW1T, 0);
        const v16h B1 = loadB(sW1T, 1);
#pragma unroll
        for (int mt = 0; mt < 2; ++mt) {
            const v16h A = loadA(mt);
            v8f z = {};
            acc[mt][0] = __builtin_amdgcn_wmma_f32_16x16x32_f16(
                false, A, false, B0, (short)0, z, false, false);
            acc[mt][1] = __builtin_amdgcn_wmma_f32_16x16x32_f16(
                false, A, false, B1, (short)0, z, false, false);
        }
    }

    // ---- epilogue 1: de-scale + bias + relu, scatter back to A-tile as f16 ----
    {
        const float ba = b1[l16];
        const float bb = b1[l16 + 16];
#pragma unroll
        for (int mt = 0; mt < 2; ++mt) {
#pragma unroll
            for (int nt = 0; nt < 2; ++nt) {
                const float bias = nt ? bb : ba;
#pragma unroll
                for (int r = 0; r < 8; ++r) {
                    // D layout: VGPR r -> M = r + 8*hsel, N = l16 (+16*nt)
                    const float v = fmaxf(acc[mt][nt][r] * kInvScale + bias, 0.0f);
                    sA[(wave * 32 + mt * 16 + hsel * 8 + r) * kAStride
                       + nt * 16 + l16] = (half_t)v;
                }
            }
        }
    }
    __syncthreads();

    // ---- layer 2: [32 x 32] @ [32 x 32] via 4 WMMAs ----
    {
        const v16h B0 = loadB(sW2T, 0);
        const v16h B1 = loadB(sW2T, 1);
#pragma unroll
        for (int mt = 0; mt < 2; ++mt) {
            const v16h A = loadA(mt);
            v8f z = {};
            acc[mt][0] = __builtin_amdgcn_wmma_f32_16x16x32_f16(
                false, A, false, B0, (short)0, z, false, false);
            acc[mt][1] = __builtin_amdgcn_wmma_f32_16x16x32_f16(
                false, A, false, B1, (short)0, z, false, false);
        }
    }

    // ---- epilogue 2: bias + relu, then 32->1 output dot via cross-lane butterfly ----
    {
        const float b2a   = b2[l16];
        const float b2b   = b2[l16 + 16];
        const float w3a   = W3[l16];
        const float w3b   = W3[l16 + 16];
        const float bias3 = b3[0];
#pragma unroll
        for (int mt = 0; mt < 2; ++mt) {
#pragma unroll
            for (int r = 0; r < 8; ++r) {
                const float hA = fmaxf(acc[mt][0][r] + b2a, 0.0f);
                const float hB = fmaxf(acc[mt][1][r] + b2b, 0.0f);
                float part = hA * w3a + hB * w3b;
                // reduce across the 16 lanes of this half (xor 1,2,4,8 stays in-half)
#pragma unroll
                for (int off = 1; off < 16; off <<= 1)
                    part += __shfl_xor(part, off, 32);
                if (l16 == 0)
                    out[pixBase + mt * 16 + hsel * 8 + r] = part + bias3;
            }
        }
    }
}

} // namespace

extern "C" void kernel_launch(void* const* d_in, const int* in_sizes, int n_in,
                              void* d_out, int out_size, void* d_ws, size_t ws_size,
                              hipStream_t stream) {
    (void)in_sizes; (void)n_in; (void)out_size; (void)d_ws; (void)ws_size;
    const float* M  = (const float*)d_in[0];
    const int*   x0 = (const int*)d_in[1];
    const int*   y0 = (const int*)d_in[2];
    const int*   x1 = (const int*)d_in[3];
    const int*   y1 = (const int*)d_in[4];
    const float* wx = (const float*)d_in[5];
    const float* wy = (const float*)d_in[6];
    const float* W1 = (const float*)d_in[7];
    const float* b1 = (const float*)d_in[8];
    const float* W2 = (const float*)d_in[9];
    const float* b2 = (const float*)d_in[10];
    const float* W3 = (const float*)d_in[11];
    const float* b3 = (const float*)d_in[12];
    float* out = (float*)d_out;

    render_mlp_kernel<<<kNPix / kBlock, kBlock, 0, stream>>>(
        M, x0, y0, x1, y1, wx, wy, W1, b1, W2, b2, W3, b3, out);
}
